// MoE_82274393522257
// MI455X (gfx1250) — matrix operand order repaired and checked by reading, not compiled, not run on hardware
//
#include <hip/hip_runtime.h>
#include <cstdint>
#include <cstddef>

// ---------------- problem sizes ----------------
#define BB 2
#define SS 2048
#define DD 2048
#define EE 8
#define FF 4096
#define NTOK (BB * SS)          // 4096 tokens
#define NSLOT_PAD 8320          // sum over experts of ceil(count/16)*16 <= 8192+8*15=8312
#define MTILES 520              // NSLOT_PAD / 16
#define OUT_N (NTOK * DD + 1)   // out tensor + aux scalar

// ---------------- WMMA vector types (wave32, 16x16x32 bf16) ----------------
typedef __attribute__((ext_vector_type(16))) __bf16 v16bf;
typedef __attribute__((ext_vector_type(4)))  __bf16 v4bf;
typedef __attribute__((ext_vector_type(8)))  float  v8f;

// K-chunk staged per iteration (two 16x16x32 WMMA sub-steps)
#define KC 64
// LDS tile strides (padded: 36*m mod 64 is a permutation -> conflict-free frags)
#define AS_LD 72     // A: 16 rows x 64 k, stride 72 bf16 (144 B)
#define BS_LD 72     // B (transposed): 256 n-rows x 64 k, stride 72 bf16 (144 B)

// ---------------------------------------------------------------------------
// WMMA fragment layouts (wave32, 16x16x32 bf16):
// A (16x32): lane L: row m=L&15, half=L>>4; elem j -> k = 16*half + j
//            => one contiguous 16-elem run per lane  -> 2 x ds_load_b128
// B (32x16): lane L: col n=L&15, khalf=L>>4; elem j -> k = 16*khalf + j
//            with k-contiguous LDS layout Bs[n][k] -> contiguous 16-elem run
//            per lane -> 2 x ds_load_b128, no packing VALU
// C (16x16 f32): lane L: col n=L&15; elem r -> row m = r + 8*(L>>4)
// Each wave owns 32 output columns; each staged KC=64 chunk = 4 WMMAs/wave.
// ---------------------------------------------------------------------------
__device__ __forceinline__ void gemm_chunk(const __bf16* __restrict__ As,
                                           const __bf16* __restrict__ Bs,
                                           int lane, int w, v8f& acc0, v8f& acc1) {
    const int m    = lane & 15;
    const int half = lane >> 4;
    const int col  = w * 32 + (lane & 15);
    const int kh   = (lane >> 4) * 16;
#pragma unroll
    for (int s = 0; s < 2; ++s) {
        const int ko = s * 32;
        v16bf a;
#pragma unroll
        for (int j = 0; j < 16; ++j) {
            const int kk = j + 16 * half;
            a[j] = As[m * AS_LD + ko + kk];
        }
        v16bf b0, b1;
#pragma unroll
        for (int j = 0; j < 16; ++j) {
            b0[j] = Bs[col * BS_LD + ko + kh + j];
            b1[j] = Bs[(col + 16) * BS_LD + ko + kh + j];
        }
        acc0 = __builtin_amdgcn_wmma_f32_16x16x32_bf16(false, a, false, b0, (short)0, acc0, false, false);
        acc1 = __builtin_amdgcn_wmma_f32_16x16x32_bf16(false, a, false, b1, (short)0, acc1, false, false);
    }
}

// Stage a 64k x 256n fp32 weight block -> bf16 LDS, TRANSPOSED to Bs[n][k].
// Each thread: four 4k x 4n micro-tiles; 4 coalesced float4 row loads each,
// in-register 4x4 transpose (v_cvt_pk_bf16_f32), 4 aligned 8-byte LDS stores.
__device__ __forceinline__ void stage_b_t(const float* __restrict__ Bsrc, size_t ldb,
                                          int k0, int n_base, int tid,
                                          __bf16* __restrict__ Bs, bool prefetch_next) {
#pragma unroll
    for (int it = 0; it < 4; ++it) {
        const int mt = tid + it * 256;        // micro-tile id (1024 total)
        const int kb = (mt >> 6) * 4;         // k offset within chunk: 0..60
        const int nb = (mt & 63) * 4;         // n offset within tile:  0..252
        const float* src = Bsrc + (size_t)(k0 + kb) * ldb + n_base + nb;
        const float4 r0 = *(const float4*)(src);
        const float4 r1 = *(const float4*)(src + ldb);
        const float4 r2 = *(const float4*)(src + 2 * ldb);
        const float4 r3 = *(const float4*)(src + 3 * ldb);
        if (prefetch_next)
            __builtin_prefetch(src + (size_t)KC * ldb, 0, 1);
        v4bf t0; t0[0]=(__bf16)r0.x; t0[1]=(__bf16)r1.x; t0[2]=(__bf16)r2.x; t0[3]=(__bf16)r3.x;
        v4bf t1; t1[0]=(__bf16)r0.y; t1[1]=(__bf16)r1.y; t1[2]=(__bf16)r2.y; t1[3]=(__bf16)r3.y;
        v4bf t2; t2[0]=(__bf16)r0.z; t2[1]=(__bf16)r1.z; t2[2]=(__bf16)r2.z; t2[3]=(__bf16)r3.z;
        v4bf t3; t3[0]=(__bf16)r0.w; t3[1]=(__bf16)r1.w; t3[2]=(__bf16)r2.w; t3[3]=(__bf16)r3.w;
        *(v4bf*)(Bs + (nb + 0) * BS_LD + kb) = t0;
        *(v4bf*)(Bs + (nb + 1) * BS_LD + kb) = t1;
        *(v4bf*)(Bs + (nb + 2) * BS_LD + kb) = t2;
        *(v4bf*)(Bs + (nb + 3) * BS_LD + kb) = t3;
    }
}

// A staging: thread owns row r = tid>>4, k-quad c = (tid&15)*4 (16x64 tile).
__device__ __forceinline__ void stage_a_zero(int tid, __bf16* __restrict__ As) {
    const int r = tid >> 4, c = (tid & 15) * 4;
    v4bf z; z[0] = (__bf16)0.f; z[1] = (__bf16)0.f; z[2] = (__bf16)0.f; z[3] = (__bf16)0.f;
    *(v4bf*)(As + r * AS_LD + c) = z;
}

__device__ __forceinline__ void stage_a_bf16(const __bf16* __restrict__ src, size_t lda,
                                             int k0, int tid, __bf16* __restrict__ As,
                                             bool rowok) {
    const int r = tid >> 4, c = (tid & 15) * 4;
    if (rowok)
        *(v4bf*)(As + r * AS_LD + c) = *(const v4bf*)(src + (size_t)r * lda + k0 + c);
}

__device__ __forceinline__ void stage_a_f32(const float* __restrict__ xrow /*row r's base*/,
                                            int k0, int tid, __bf16* __restrict__ As,
                                            bool rowok) {
    const int r = tid >> 4, c = (tid & 15) * 4;
    if (rowok) {
        const float4 v = *(const float4*)(xrow + k0 + c);
        v4bf t; t[0] = (__bf16)v.x; t[1] = (__bf16)v.y; t[2] = (__bf16)v.z; t[3] = (__bf16)v.w;
        *(v4bf*)(As + r * AS_LD + c) = t;
    }
}

// ---------------------------------------------------------------------------
// K0: zero out + counters
// ---------------------------------------------------------------------------
__global__ __launch_bounds__(256) void init_kernel(float* __restrict__ out, int n,
                                                   int* __restrict__ counts) {
    const int i = blockIdx.x * 256 + threadIdx.x;
    if (i < n) out[i] = 0.0f;
    if (i < EE) counts[i] = 0;
}

// ---------------------------------------------------------------------------
// K1: LayerNorm -> bf16 x_norm   (one block per token)
// ---------------------------------------------------------------------------
__global__ __launch_bounds__(256) void ln_kernel(const float* __restrict__ x,
                                                 const float* __restrict__ gamma,
                                                 const float* __restrict__ beta,
                                                 __bf16* __restrict__ xnorm) {
    __shared__ float rs[256], rq[256];
    __shared__ float stats[2];
    const int tok = blockIdx.x, tid = threadIdx.x;
    const float* xr = x + (size_t)tok * DD;
    float v[8], s = 0.f, q = 0.f;
#pragma unroll
    for (int j = 0; j < 8; ++j) {
        const float t = xr[tid + j * 256];
        v[j] = t; s += t; q += t * t;
    }
    rs[tid] = s; rq[tid] = q; __syncthreads();
    for (int off = 128; off > 0; off >>= 1) {
        if (tid < off) { rs[tid] += rs[tid + off]; rq[tid] += rq[tid + off]; }
        __syncthreads();
    }
    if (tid == 0) {
        const float mu  = rs[0] * (1.0f / DD);
        const float var = rq[0] * (1.0f / DD) - mu * mu;
        stats[0] = mu; stats[1] = rsqrtf(var + 1e-5f);
    }
    __syncthreads();
    const float mu = stats[0], rstd = stats[1];
    __bf16* xo = xnorm + (size_t)tok * DD;
#pragma unroll
    for (int j = 0; j < 8; ++j) {
        const int d = tid + j * 256;
        xo[d] = (__bf16)((v[j] - mu) * rstd * gamma[d] + beta[d]);
    }
}

// ---------------------------------------------------------------------------
// K2: router MLP1: rh = relu(xnorm @ rW1 + rb1)   [4096 x 2048], bf16 WMMA
// grid (256 m-tiles, 8 n-tiles), 256 threads = 8 waves, wave owns 32 cols
// ---------------------------------------------------------------------------
__global__ __launch_bounds__(256) void router_gemm_kernel(const __bf16* __restrict__ xnorm,
                                                          const float* __restrict__ rW1,
                                                          const float* __restrict__ rb1,
                                                          __bf16* __restrict__ rh) {
    __shared__ __bf16 As[16 * AS_LD];
    __shared__ __bf16 Bs[256 * BS_LD];
    const int mbase  = blockIdx.x * 16;
    const int n_base = blockIdx.y * 256;
    const int tid = threadIdx.x, lane = tid & 31, w = tid >> 5;
    const __bf16* Arow = xnorm + (size_t)mbase * DD;
    v8f acc0 = {0.f,0.f,0.f,0.f,0.f,0.f,0.f,0.f};
    v8f acc1 = {0.f,0.f,0.f,0.f,0.f,0.f,0.f,0.f};
    for (int k0 = 0; k0 < DD; k0 += KC) {
        stage_a_bf16(Arow, DD, k0, tid, As, true);
        stage_b_t(rW1, DD, k0, n_base, tid, Bs, (k0 + KC) < DD);
        __syncthreads();
        gemm_chunk(As, Bs, lane, w, acc0, acc1);
        __syncthreads();
    }
    const int n0 = lane & 15, half = lane >> 4;
#pragma unroll
    for (int r = 0; r < 8; ++r) {
        const int m  = r + 8 * half;
        const int f0 = n_base + w * 32 + n0;
        const float v0 = fmaxf(acc0[r] + rb1[f0], 0.f);
        const float v1 = fmaxf(acc1[r] + rb1[f0 + 16], 0.f);
        __bf16* dst = rh + (size_t)(mbase + m) * DD;
        dst[f0]      = (__bf16)v0;
        dst[f0 + 16] = (__bf16)v1;
    }
}

// ---------------------------------------------------------------------------
// K3: logits = rh @ rW2 + rb2 ; softmax ; top-2 ; per-token record (1 wave/token)
// ---------------------------------------------------------------------------
__global__ __launch_bounds__(256) void router_top2_kernel(const __bf16* __restrict__ rh,
                                                          const float* __restrict__ rW2,
                                                          const float* __restrict__ rb2,
                                                          float* __restrict__ probs,
                                                          int* __restrict__ top_e,
                                                          float* __restrict__ top_p,
                                                          int* __restrict__ counts) {
    const int wid  = blockIdx.x * 8 + (threadIdx.x >> 5);   // token id
    const int lane = threadIdx.x & 31;
    const __bf16* hrow = rh + (size_t)wid * DD;
    float acc[EE] = {0.f,0.f,0.f,0.f,0.f,0.f,0.f,0.f};
    for (int d = lane; d < DD; d += 32) {
        const float hv = (float)hrow[d];
        const float4* wp = (const float4*)(rW2 + d * EE);
        const float4 w0 = wp[0], w1 = wp[1];
        acc[0] += hv * w0.x; acc[1] += hv * w0.y; acc[2] += hv * w0.z; acc[3] += hv * w0.w;
        acc[4] += hv * w1.x; acc[5] += hv * w1.y; acc[6] += hv * w1.z; acc[7] += hv * w1.w;
    }
#pragma unroll
    for (int e = 0; e < EE; ++e)
        for (int off = 16; off > 0; off >>= 1) acc[e] += __shfl_xor(acc[e], off);
    // all lanes hold full sums now
    float l[EE]; float mx = -1e30f;
#pragma unroll
    for (int e = 0; e < EE; ++e) { l[e] = acc[e] + rb2[e]; mx = fmaxf(mx, l[e]); }
    float s = 0.f;
#pragma unroll
    for (int e = 0; e < EE; ++e) { l[e] = expf(l[e] - mx); s += l[e]; }
    const float inv = 1.0f / s;
    int i0 = 0; float p0 = -1.f;
#pragma unroll
    for (int e = 0; e < EE; ++e) { l[e] *= inv; if (l[e] > p0) { p0 = l[e]; i0 = e; } }
    int i1 = 0; float p1 = -1.f;
#pragma unroll
    for (int e = 0; e < EE; ++e) { if (e != i0 && l[e] > p1) { p1 = l[e]; i1 = e; } }
    const float pn = p0 + p1;
    if (lane == 0) {
#pragma unroll
        for (int e = 0; e < EE; ++e) probs[wid * EE + e] = l[e];
        top_e[wid * 2]     = i0;       top_e[wid * 2 + 1] = i1;
        top_p[wid * 2]     = p0 / pn;  top_p[wid * 2 + 1] = p1 / pn;
        atomicAdd(&counts[i0], 1);
        atomicAdd(&counts[i1], 1);
    }
}

// ---------------------------------------------------------------------------
// K4: deterministic aux-loss reduction + 16-aligned expert offsets/cursors
// ---------------------------------------------------------------------------
__global__ __launch_bounds__(256) void offsets_aux_kernel(const float* __restrict__ probs,
                                                          const int* __restrict__ counts,
                                                          int* __restrict__ offs,
                                                          int* __restrict__ cursor,
                                                          float* __restrict__ out) {
    __shared__ float red[256];
    __shared__ float psum[EE];
    const int tid = threadIdx.x;
    float s[EE] = {0.f,0.f,0.f,0.f,0.f,0.f,0.f,0.f};
    for (int t = tid; t < NTOK; t += 256) {
        const float* p = probs + t * EE;
#pragma unroll
        for (int e = 0; e < EE; ++e) s[e] += p[e];
    }
    for (int e = 0; e < EE; ++e) {
        red[tid] = s[e]; __syncthreads();
        for (int off = 128; off > 0; off >>= 1) {
            if (tid < off) red[tid] += red[tid + off];
            __syncthreads();
        }
        if (tid == 0) psum[e] = red[0];
        __syncthreads();
    }
    if (tid == 0) {
        float aux = 0.f;
#pragma unroll
        for (int e = 0; e < EE; ++e) {
            const float pm = psum[e] * (1.0f / NTOK);
            aux += pm * logf(pm * (float)EE + 1e-9f);
        }
        out[OUT_N - 1] = aux;
        int run = 0;
#pragma unroll
        for (int e = 0; e < EE; ++e) {
            offs[e] = run; cursor[e] = run;
            run += (counts[e] + 15) & ~15;
        }
        offs[EE] = run;
    }
}

// ---------------------------------------------------------------------------
// K5: compact token slots into per-expert 16-aligned regions
// ---------------------------------------------------------------------------
__global__ __launch_bounds__(256) void compact_kernel(const int* __restrict__ top_e,
                                                      const float* __restrict__ top_p,
                                                      int* __restrict__ cursor,
                                                      int* __restrict__ tok_arr,
                                                      float* __restrict__ p_arr) {
    const int tok = blockIdx.x * 256 + threadIdx.x;   // 16 blocks * 256 = 4096
#pragma unroll
    for (int sidx = 0; sidx < 2; ++sidx) {
        const int e   = top_e[tok * 2 + sidx];
        const int pos = atomicAdd(&cursor[e], 1);
        tok_arr[pos] = tok;
        p_arr[pos]   = top_p[tok * 2 + sidx];
    }
}

// ---------------------------------------------------------------------------
// K6: expert fc1 (grouped GEMM): h = relu(x[gathered] @ We1[e] + be1[e]) -> bf16
// grid (520 m-tiles, 16 n-tiles of F)
// ---------------------------------------------------------------------------
__global__ __launch_bounds__(256) void expert_fc1_kernel(const float* __restrict__ x,
                                                         const float* __restrict__ We1,
                                                         const float* __restrict__ be1,
                                                         const int* __restrict__ counts,
                                                         const int* __restrict__ offs,
                                                         const int* __restrict__ tok_arr,
                                                         __bf16* __restrict__ hbuf) {
    const int base16 = blockIdx.x * 16;
    if (base16 >= offs[EE]) return;
    int e = 0;
    while (e < EE - 1 && base16 >= offs[e + 1]) ++e;
    const int local = base16 - offs[e];
    const int cnt   = counts[e];
    if (local >= cnt) return;
    const int n_base = blockIdx.y * 256;
    const int tid = threadIdx.x, lane = tid & 31, w = tid >> 5;

    __shared__ __bf16 As[16 * AS_LD];
    __shared__ __bf16 Bs[256 * BS_LD];

    const float* Bsrc = We1 + (size_t)e * DD * FF;
    // gathered A rows: thread owns row r = tid>>4 of the stage tile
    const int  ar = tid >> 4;
    const bool rowok = (local + ar) < cnt;
    const int  tok = rowok ? tok_arr[base16 + ar] : 0;
    const float* xrow = x + (size_t)tok * DD;

    stage_a_zero(tid, As);   // rows beyond count stay zero for the whole kernel

    v8f acc0 = {0.f,0.f,0.f,0.f,0.f,0.f,0.f,0.f};
    v8f acc1 = {0.f,0.f,0.f,0.f,0.f,0.f,0.f,0.f};
    for (int k0 = 0; k0 < DD; k0 += KC) {
        stage_a_f32(xrow, k0, tid, As, rowok);
        stage_b_t(Bsrc, FF, k0, n_base, tid, Bs, (k0 + KC) < DD);
        __syncthreads();
        gemm_chunk(As, Bs, lane, w, acc0, acc1);
        __syncthreads();
    }
    const int n0 = lane & 15, half = lane >> 4;
    const float* be1e = be1 + (size_t)e * FF;
#pragma unroll
    for (int r = 0; r < 8; ++r) {
        const int m = r + 8 * half;
        if (local + m < cnt) {
            const int f0 = n_base + w * 32 + n0;
            const float v0 = fmaxf(acc0[r] + be1e[f0], 0.f);
            const float v1 = fmaxf(acc1[r] + be1e[f0 + 16], 0.f);
            __bf16* hr = hbuf + (size_t)(base16 + m) * FF;
            hr[f0]      = (__bf16)v0;
            hr[f0 + 16] = (__bf16)v1;
        }
    }
}

// ---------------------------------------------------------------------------
// K7: expert fc2 + weighted combine: out[tok] += p * (h @ We2[e] + be2[e])
// grid (520 m-tiles, 8 n-tiles of D); exactly 2 atomic adds per output element
// (0+a)+b == (0+b)+a bitwise, so the result is replay-deterministic.
// ---------------------------------------------------------------------------
__global__ __launch_bounds__(256) void expert_fc2_kernel(const __bf16* __restrict__ hbuf,
                                                         const float* __restrict__ We2,
                                                         const float* __restrict__ be2,
                                                         const int* __restrict__ counts,
                                                         const int* __restrict__ offs,
                                                         const int* __restrict__ tok_arr,
                                                         const float* __restrict__ p_arr,
                                                         float* __restrict__ out) {
    const int base16 = blockIdx.x * 16;
    if (base16 >= offs[EE]) return;
    int e = 0;
    while (e < EE - 1 && base16 >= offs[e + 1]) ++e;
    const int local = base16 - offs[e];
    const int cnt   = counts[e];
    if (local >= cnt) return;
    const int n_base = blockIdx.y * 256;
    const int tid = threadIdx.x, lane = tid & 31, w = tid >> 5;

    __shared__ __bf16 As[16 * AS_LD];
    __shared__ __bf16 Bs[256 * BS_LD];

    const float*  Bsrc = We2 + (size_t)e * FF * DD;
    const __bf16* Arow = hbuf + (size_t)base16 * FF;
    const bool rowok = (local + (tid >> 4)) < cnt;

    stage_a_zero(tid, As);

    v8f acc0 = {0.f,0.f,0.f,0.f,0.f,0.f,0.f,0.f};
    v8f acc1 = {0.f,0.f,0.f,0.f,0.f,0.f,0.f,0.f};
    for (int k0 = 0; k0 < FF; k0 += KC) {
        stage_a_bf16(Arow, FF, k0, tid, As, rowok);
        stage_b_t(Bsrc, DD, k0, n_base, tid, Bs, (k0 + KC) < FF);
        __syncthreads();
        gemm_chunk(As, Bs, lane, w, acc0, acc1);
        __syncthreads();
    }
    const int n0 = lane & 15, half = lane >> 4;
    const float* be2e = be2 + (size_t)e * DD;
#pragma unroll
    for (int r = 0; r < 8; ++r) {
        const int m = r + 8 * half;
        if (local + m < cnt) {
            const int   tok = tok_arr[base16 + m];
            const float p   = p_arr[base16 + m];
            const int   d0  = n_base + w * 32 + n0;
            const float v0  = acc0[r] + be2e[d0];
            const float v1  = acc1[r] + be2e[d0 + 16];
            atomicAdd(out + (size_t)tok * DD + d0,      p * v0);
            atomicAdd(out + (size_t)tok * DD + d0 + 16, p * v1);
        }
    }
}

// ---------------------------------------------------------------------------
// host-side launch.  Workspace layout (~97 MiB).
// ---------------------------------------------------------------------------
extern "C" void kernel_launch(void* const* d_in, const int* in_sizes, int n_in,
                              void* d_out, int out_size, void* d_ws, size_t ws_size,
                              hipStream_t stream) {
    const float* x     = (const float*)d_in[0];
    const float* gamma = (const float*)d_in[1];
    const float* beta  = (const float*)d_in[2];
    const float* rW1   = (const float*)d_in[3];
    const float* rb1   = (const float*)d_in[4];
    const float* rW2   = (const float*)d_in[5];
    const float* rb2   = (const float*)d_in[6];
    const float* We1   = (const float*)d_in[7];
    const float* be1   = (const float*)d_in[8];
    const float* We2   = (const float*)d_in[9];
    const float* be2   = (const float*)d_in[10];
    float* out = (float*)d_out;

    char* ws = (char*)d_ws;
    constexpr size_t WS_COUNTS = 0;                         // int[8]
    constexpr size_t WS_OFFS   = 64;                        // int[9]
    constexpr size_t WS_CURSOR = 128;                       // int[8]
    constexpr size_t WS_TOPE   = 256;                       // int[2*4096]
    constexpr size_t WS_TOPP   = WS_TOPE + 32768;           // float[2*4096]
    constexpr size_t WS_PROBS  = WS_TOPP + 32768;           // float[4096*8]
    constexpr size_t WS_TOK    = WS_PROBS + 131072;         // int[8320]
    constexpr size_t WS_PARR   = WS_TOK + 33280;            // float[8320]
    constexpr size_t WS_XNORM  = 263424;                    // bf16[4096*2048]
    constexpr size_t WS_RH     = WS_XNORM + 16777216;       // bf16[4096*2048]
    constexpr size_t WS_H      = WS_RH + 16777216;          // bf16[8320*4096]

    int*    counts  = (int*)(ws + WS_COUNTS);
    int*    offs    = (int*)(ws + WS_OFFS);
    int*    cursor  = (int*)(ws + WS_CURSOR);
    int*    top_e   = (int*)(ws + WS_TOPE);
    float*  top_p   = (float*)(ws + WS_TOPP);
    float*  probs   = (float*)(ws + WS_PROBS);
    int*    tok_arr = (int*)(ws + WS_TOK);
    float*  p_arr   = (float*)(ws + WS_PARR);
    __bf16* xnorm   = (__bf16*)(ws + WS_XNORM);
    __bf16* rh      = (__bf16*)(ws + WS_RH);
    __bf16* hbuf    = (__bf16*)(ws + WS_H);

    const int zgrid = (OUT_N + 255) / 256;
    init_kernel<<<zgrid, 256, 0, stream>>>(out, OUT_N, counts);
    ln_kernel<<<NTOK, 256, 0, stream>>>(x, gamma, beta, xnorm);
    router_gemm_kernel<<<dim3(NTOK / 16, DD / 256), 256, 0, stream>>>(xnorm, rW1, rb1, rh);
    router_top2_kernel<<<NTOK / 8, 256, 0, stream>>>(rh, rW2, rb2, probs, top_e, top_p, counts);
    offsets_aux_kernel<<<1, 256, 0, stream>>>(probs, counts, offs, cursor, out);
    compact_kernel<<<NTOK / 256, 256, 0, stream>>>(top_e, top_p, cursor, tok_arr, p_arr);
    expert_fc1_kernel<<<dim3(MTILES, FF / 256), 256, 0, stream>>>(x, We1, be1, counts, offs,
                                                                  tok_arr, hbuf);
    expert_fc2_kernel<<<dim3(MTILES, DD / 256), 256, 0, stream>>>(hbuf, We2, be2, counts, offs,
                                                                  tok_arr, p_arr, out);
}
